// TransformerBlock_54150947668208
// MI455X (gfx1250) — compile-verified, matrix operand order checked by reading, not run on hardware
//
#include <hip/hip_runtime.h>

// ---------------------------------------------------------------------------
// Types for CDNA5 WMMA (gfx1250, wave32)
// ---------------------------------------------------------------------------
typedef __bf16 bf16;
typedef __attribute__((ext_vector_type(8)))  bf16  v8bf;
typedef __attribute__((ext_vector_type(16))) bf16  v16bf;
typedef __attribute__((ext_vector_type(8)))  float v8f;

#define HW    16384
#define WIMG  128
#define C192  192
#define NB    2
#define BIG   1020
#define BIGP  1024
#define HALF_BIG 510

// A/B fragment loader for v_wmma_f32_16x16x32_bf16.
// Per ISA 7.12.2 (16-bit A 16x32): lane<16 holds k0..k0+7 (elems 0..7) and
// k0+16..k0+23 (elems 8..15); lane>=16 holds the +8 shifted K ranges.
// Both chunks are 8 contiguous bf16 = one 16B load each.
__device__ __forceinline__ v16bf frag_load(const bf16* __restrict__ row, int k0, int hiSel) {
  const v8bf lo = *reinterpret_cast<const v8bf*>(row + k0 + hiSel);
  const v8bf hi = *reinterpret_cast<const v8bf*>(row + k0 + 16 + hiSel);
  return __builtin_shufflevector(lo, hi, 0,1,2,3,4,5,6,7,8,9,10,11,12,13,14,15);
}

__device__ __forceinline__ v8f wmma_bf16(v16bf a, v16bf b, v8f c) {
  return __builtin_amdgcn_wmma_f32_16x16x32_bf16(false, a, false, b, (short)0, c, false, false);
}

__device__ __forceinline__ v8f v8f_zero() {
  v8f z = {0.f,0.f,0.f,0.f,0.f,0.f,0.f,0.f};
  return z;
}

// ---------------------------------------------------------------------------
// LayerNorm kernels (per-pixel over 192 channels)
// ---------------------------------------------------------------------------
__global__ void ln_dual_kernel(const float* __restrict__ in,
                               const float* __restrict__ w1, const float* __restrict__ b1,
                               const float* __restrict__ w2, const float* __restrict__ b2,
                               float* __restrict__ o1, float* __restrict__ o2) {
  int g = blockIdx.x * blockDim.x + threadIdx.x;   // 0 .. NB*HW-1
  if (g >= NB * HW) return;
  int b = g >> 14, p = g & (HW - 1);
  const float* base = in + (size_t)b * C192 * HW + p;
  float s = 0.f, ss = 0.f;
  for (int c = 0; c < C192; ++c) { float x = base[(size_t)c * HW]; s += x; ss += x * x; }
  float mu  = s * (1.f / C192);
  float var = ss * (1.f / C192) - mu * mu;
  float inv = rsqrtf(var + 1e-5f);
  float* p1 = o1 + (size_t)b * C192 * HW + p;
  float* p2 = o2 ? o2 + (size_t)b * C192 * HW + p : nullptr;
  for (int c = 0; c < C192; ++c) {
    float xn = (base[(size_t)c * HW] - mu) * inv;
    p1[(size_t)c * HW] = xn * w1[c] + b1[c];
    if (p2) p2[(size_t)c * HW] = xn * w2[c] + b2[c];
  }
}

// LN writing transposed bf16 (N, 192) for GEMM B-operand consumption.
__global__ void ln_t_kernel(const float* __restrict__ in,
                            const float* __restrict__ w, const float* __restrict__ b,
                            bf16* __restrict__ yt) {
  int g = blockIdx.x * blockDim.x + threadIdx.x;
  if (g >= NB * HW) return;
  int bb = g >> 14, p = g & (HW - 1);
  const float* base = in + (size_t)bb * C192 * HW + p;
  float s = 0.f, ss = 0.f;
  for (int c = 0; c < C192; ++c) { float x = base[(size_t)c * HW]; s += x; ss += x * x; }
  float mu  = s * (1.f / C192);
  float var = ss * (1.f / C192) - mu * mu;
  float inv = rsqrtf(var + 1e-5f);
  bf16* dst = yt + (size_t)g * C192;
  for (int c = 0; c < C192; ++c) {
    float xn = (base[(size_t)c * HW] - mu) * inv;
    dst[c] = (bf16)(xn * w[c] + b[c]);
  }
}

// ---------------------------------------------------------------------------
// Depthwise / grouped convolutions
// ---------------------------------------------------------------------------
template<int KS, typename T>
__device__ __forceinline__ float conv_at(const T* __restrict__ ch,
                                         const float* __restrict__ w, int y, int x) {
  constexpr int P = KS / 2;
  float acc = 0.f;
#pragma unroll
  for (int ky = 0; ky < KS; ++ky) {
    int yy = y + ky - P;
    if (yy < 0 || yy >= WIMG) continue;
#pragma unroll
    for (int kx = 0; kx < KS; ++kx) {
      int xx = x + kx - P;
      if (xx < 0 || xx >= WIMG) continue;
      acc += w[ky * KS + kx] * (float)ch[yy * WIMG + xx];
    }
  }
  return acc;
}

// Depthwise 3x3, fp32 out + per-(b,c) block-partial sum of squares (for l2norm).
__global__ void dw3_f32_ssq(const float* __restrict__ in, const float* __restrict__ wgt,
                            float* __restrict__ out, float* __restrict__ ssq_part) {
  int p = blockIdx.x * blockDim.x + threadIdx.x;
  int c = blockIdx.y, b = blockIdx.z;
  const float* ch = in + ((size_t)b * C192 + c) * HW;
  float acc = conv_at<3>(ch, wgt + c * 9, p >> 7, p & 127);
  out[((size_t)b * C192 + c) * HW + p] = acc;
  __shared__ float red[256];
  red[threadIdx.x] = acc * acc;
  __syncthreads();
  for (int s = 128; s > 0; s >>= 1) {
    if (threadIdx.x < s) red[threadIdx.x] += red[threadIdx.x + s];
    __syncthreads();
  }
  if (threadIdx.x == 0) ssq_part[((size_t)b * C192 + c) * 64 + blockIdx.x] = red[0];
}

// Depthwise 3x3 for v, written transposed bf16 into v_t[b][n][h*64 + d] (d padded 48->64).
__global__ void dw3_vt(const float* __restrict__ in, const float* __restrict__ wgt,
                       bf16* __restrict__ vt) {
  int p = blockIdx.x * blockDim.x + threadIdx.x;
  int c = blockIdx.y, b = blockIdx.z;
  const float* ch = in + ((size_t)b * C192 + c) * HW;
  float acc = conv_at<3>(ch, wgt + c * 9, p >> 7, p & 127);
  int col = (c / 48) * 64 + (c % 48);
  vt[((size_t)b * HW + p) * 256 + col] = (bf16)acc;
}

// Depthwise KSxKS on 1020-channel bf16 tensor, ReLU, bf16 (c,n) out.
template<int KS>
__global__ void dw_relu_bf16(const bf16* __restrict__ in, const float* __restrict__ wgt,
                             bf16* __restrict__ out) {
  int p = blockIdx.x * blockDim.x + threadIdx.x;
  int c = blockIdx.y, b = blockIdx.z;
  const bf16* ch = in + ((size_t)b * BIG + c) * HW;
  float acc = conv_at<KS>(ch, wgt + c * KS * KS, p >> 7, p & 127);
  out[((size_t)b * BIG + c) * HW + p] = (bf16)fmaxf(acc, 0.f);
}

// Grouped conv (groups=510, 2-in -> 1-out), input channels resolved from the
// x1/x2 = concat(split(x3), split(x5)) mapping. ReLU, transposed bf16 out.
template<int KS>
__global__ void grp2_bf16(const bf16* __restrict__ x3, const bf16* __restrict__ x5,
                          const float* __restrict__ wgt, bf16* __restrict__ ffnt,
                          int which) {
  int p = blockIdx.x * blockDim.x + threadIdx.x;
  int g = blockIdx.y, b = blockIdx.z;
  int y = p >> 7, x = p & 127;
  int j0 = 2 * g, j1 = 2 * g + 1;
  const bf16* chA;
  const bf16* chB;
  if (which == 0) {  // x1: [x3[0:510], x5[0:510]]
    chA = (j0 < HALF_BIG) ? x3 + ((size_t)b * BIG + j0) * HW
                          : x5 + ((size_t)b * BIG + (j0 - HALF_BIG)) * HW;
    chB = (j1 < HALF_BIG) ? x3 + ((size_t)b * BIG + j1) * HW
                          : x5 + ((size_t)b * BIG + (j1 - HALF_BIG)) * HW;
  } else {           // x2: [x3[510:1020], x5[510:1020]]
    chA = (j0 < HALF_BIG) ? x3 + ((size_t)b * BIG + HALF_BIG + j0) * HW
                          : x5 + ((size_t)b * BIG + j0) * HW;
    chB = (j1 < HALF_BIG) ? x3 + ((size_t)b * BIG + HALF_BIG + j1) * HW
                          : x5 + ((size_t)b * BIG + j1) * HW;
  }
  float acc = conv_at<KS>(chA, wgt + (g * 2 + 0) * KS * KS, y, x)
            + conv_at<KS>(chB, wgt + (g * 2 + 1) * KS * KS, y, x);
  acc = fmaxf(acc, 0.f);
  int col = (which == 0) ? g : HALF_BIG + g;
  ffnt[((size_t)b * HW + p) * BIGP + col] = (bf16)acc;
}

// ---------------------------------------------------------------------------
// Small utility kernels
// ---------------------------------------------------------------------------
__global__ void reduce_inv(const float* __restrict__ part, float* __restrict__ inv, int n) {
  int i = blockIdx.x * blockDim.x + threadIdx.x;
  if (i >= n) return;
  float s = 0.f;
  for (int j = 0; j < 64; ++j) s += part[(size_t)i * 64 + j];
  inv[i] = 1.f / fmaxf(sqrtf(s), 1e-12f);
}

__global__ void scale_bf16_kernel(const float* __restrict__ x, const float* __restrict__ inv,
                                  bf16* __restrict__ o, int total) {
  int i = blockIdx.x * blockDim.x + threadIdx.x;
  if (i >= total) return;
  o[i] = (bf16)(x[i] * inv[i >> 14]);
}

__global__ void pack_w_kernel(const float* __restrict__ src, bf16* __restrict__ dst,
                              int M, int K, int Mp, int Kp) {
  int i = blockIdx.x * blockDim.x + threadIdx.x;
  if (i >= Mp * Kp) return;
  int m = i / Kp, k = i % Kp;
  dst[i] = (m < M && k < K) ? (bf16)src[(size_t)m * K + k] : (bf16)0.f;
}

__global__ void zero_vt_pad(bf16* __restrict__ vt) {
  int i = blockIdx.x * blockDim.x + threadIdx.x;      // NB*HW*64
  if (i >= NB * HW * 64) return;
  int row = i >> 6, j = i & 63;
  int h = j >> 4, d = j & 15;
  vt[(size_t)row * 256 + h * 64 + 48 + d] = (bf16)0.f;
}

__global__ void zero_ffnt_pad(bf16* __restrict__ ffnt) {
  int i = blockIdx.x * blockDim.x + threadIdx.x;      // NB*HW*4
  if (i >= NB * HW * 4) return;
  int row = i >> 2, j = i & 3;
  ffnt[(size_t)row * BIGP + BIG + j] = (bf16)0.f;
}

__global__ void reduce_attn(const float* __restrict__ part, float* __restrict__ attn) {
  int i = blockIdx.x * blockDim.x + threadIdx.x;      // 8*48*48
  if (i >= 8 * 48 * 48) return;
  float s = 0.f;
  for (int sp = 0; sp < 32; ++sp) s += part[(size_t)sp * 8 * 2304 + i];
  attn[i] = s;
}

// ---------------------------------------------------------------------------
// attn = q . k^T  (per (b,head): 48x48, K=16384), split-K WMMA, deterministic.
// ---------------------------------------------------------------------------
__global__ void attn_gemm(const bf16* __restrict__ qbf, const bf16* __restrict__ kbf,
                          float* __restrict__ attn_part) {
  int bh = blockIdx.x;                 // 0..7
  int b = bh >> 2, h = bh & 3;
  int split = blockIdx.y;              // 0..31
  int lane = threadIdx.x;              // 32 threads (one wave)
  int l15 = lane & 15;
  int hiSel = (lane >> 4) * 8;
  const bf16* qb = qbf + ((size_t)b * C192 + h * 48) * HW;
  const bf16* kb = kbf + ((size_t)b * C192 + h * 48) * HW;
  v8f acc[3][3];
#pragma unroll
  for (int i = 0; i < 3; ++i)
#pragma unroll
    for (int j = 0; j < 3; ++j) acc[i][j] = v8f_zero();

  int k0base = split * 512;
  for (int k0 = k0base; k0 < k0base + 512; k0 += 32) {
    v16bf A[3], Bf[3];
#pragma unroll
    for (int mb = 0; mb < 3; ++mb) A[mb]  = frag_load(qb + (size_t)(mb * 16 + l15) * HW, k0, hiSel);
#pragma unroll
    for (int nb = 0; nb < 3; ++nb) Bf[nb] = frag_load(kb + (size_t)(nb * 16 + l15) * HW, k0, hiSel);
#pragma unroll
    for (int mb = 0; mb < 3; ++mb)
#pragma unroll
      for (int nb = 0; nb < 3; ++nb)
        acc[mb][nb] = wmma_bf16(A[mb], Bf[nb], acc[mb][nb]);
  }
  float* dst = attn_part + (size_t)split * 8 * 2304 + (size_t)bh * 2304;
#pragma unroll
  for (int mb = 0; mb < 3; ++mb)
#pragma unroll
    for (int nb = 0; nb < 3; ++nb) {
      int n = nb * 16 + l15;
#pragma unroll
      for (int r = 0; r < 8; ++r) {
        int m = mb * 16 + hiSel + r;   // C layout: lane>=16 -> M=8..15
        dst[m * 48 + n] = acc[mb][nb][r];
      }
    }
}

// ---------------------------------------------------------------------------
// temperature + four top-k sparsified softmaxes combined into one P (bf16).
// ---------------------------------------------------------------------------
__global__ void combine_topk(const float* __restrict__ attn, const float* __restrict__ temp,
                             const float* __restrict__ a1, const float* __restrict__ a2,
                             const float* __restrict__ a3, const float* __restrict__ a4,
                             bf16* __restrict__ Pbf) {
  int bh = blockIdx.x;
  int h = bh & 3;
  __shared__ float S[48 * 48];
  float t = temp[h];
  for (int i = threadIdx.x; i < 48 * 48; i += blockDim.x)
    S[i] = attn[(size_t)bh * 2304 + i] * t;
  __syncthreads();
  int c = threadIdx.x;
  if (c < 48) {
    const float* row = S + c * 48;
    float m = -1e30f;
    for (int j = 0; j < 48; ++j) m = fmaxf(m, row[j]);
    const int KK[4] = {24, 32, 36, 38};
    float wbr[4] = {a1[0], a2[0], a3[0], a4[0]};
    float thr[4], invS[4];
#pragma unroll
    for (int br = 0; br < 4; ++br) {
      int k = KK[br];
      float th = -1e30f;
      for (int i = 0; i < 48; ++i) {
        float vi = row[i];
        int cgt = 0, ceq = 0;
        for (int j = 0; j < 48; ++j) { cgt += (row[j] > vi); ceq += (row[j] == vi); }
        if (cgt < k && cgt + ceq >= k) { th = vi; break; }   // k-th largest
      }
      thr[br] = th;
      float ssum = 0.f;
      for (int j = 0; j < 48; ++j) if (row[j] >= th) ssum += expf(row[j] - m);
      invS[br] = wbr[br] / ssum;
    }
    bf16* dst = Pbf + (size_t)bh * 48 * 64 + c * 64;
    for (int d = 0; d < 48; ++d) {
      float e = expf(row[d] - m);
      float o = 0.f;
#pragma unroll
      for (int br = 0; br < 4; ++br) if (row[d] >= thr[br]) o += invS[br] * e;
      dst[d] = (bf16)o;
    }
    for (int d = 48; d < 64; ++d) dst[d] = (bf16)0.f;
  }
}

// ---------------------------------------------------------------------------
// out = P @ v  (M=48, K=64 padded, N=16384 per (b,h)); transposed bf16 out.
// ---------------------------------------------------------------------------
__global__ void pv_gemm(const bf16* __restrict__ Pbf, const bf16* __restrict__ vt,
                        bf16* __restrict__ pot) {
  int bh = blockIdx.y;
  int b = bh >> 2, h = bh & 3;
  int wave = threadIdx.x >> 5, lane = threadIdx.x & 31;
  int l15 = lane & 15, hiSel = (lane >> 4) * 8;
  int n0 = blockIdx.x * 64 + wave * 16;
  const bf16* Pb = Pbf + (size_t)bh * 48 * 64;
  const bf16* vb = vt + (size_t)b * HW * 256 + h * 64;
  v8f acc[3];
#pragma unroll
  for (int i = 0; i < 3; ++i) acc[i] = v8f_zero();
#pragma unroll
  for (int k0 = 0; k0 < 64; k0 += 32) {
    v16bf Bf = frag_load(vb + (size_t)(n0 + l15) * 256, k0, hiSel);
#pragma unroll
    for (int mb = 0; mb < 3; ++mb) {
      v16bf A = frag_load(Pb + (size_t)(mb * 16 + l15) * 64, k0, hiSel);
      acc[mb] = wmma_bf16(A, Bf, acc[mb]);
    }
  }
  int n = n0 + l15;
  bf16* dst = pot + ((size_t)b * HW + n) * C192 + h * 48;
#pragma unroll
  for (int mb = 0; mb < 3; ++mb) {
    v8bf pk;
#pragma unroll
    for (int r = 0; r < 8; ++r) pk[r] = (bf16)acc[mb][r];
    *reinterpret_cast<v8bf*>(dst + mb * 16 + hiSel) = pk;   // 8 consecutive channels
  }
}

// ---------------------------------------------------------------------------
// Generic TN GEMM: out[m,n] = sum_k A[m,k] * Bt[n,k] (+res), WMMA bf16.
// A: (Mpad,Kp) bf16 zero-padded, Bt: (HW,Kp) bf16 per batch.
// Block 256 = 8 waves (4M x 2N); block tile 64(M) x 128(N); wave 16x64.
// ---------------------------------------------------------------------------
template<int OUTF32>
__global__ void gemm_tn(const bf16* __restrict__ A, const bf16* __restrict__ Bt,
                        int M, int Kp, const float* __restrict__ res,
                        float* __restrict__ outF, bf16* __restrict__ outB) {
  int b = blockIdx.z;
  const bf16* Btb = Bt + (size_t)b * HW * Kp;
  int wave = threadIdx.x >> 5, lane = threadIdx.x & 31;
  int mw = wave & 3, nw = wave >> 2;
  int l15 = lane & 15, hiSel = (lane >> 4) * 8;
  int mrow = blockIdx.y * 64 + mw * 16 + l15;
  const bf16* Arow = A + (size_t)mrow * Kp;
  int n0 = blockIdx.x * 128 + nw * 64;
  v8f acc[4];
#pragma unroll
  for (int i = 0; i < 4; ++i) acc[i] = v8f_zero();
  for (int k0 = 0; k0 < Kp; k0 += 32) {
    v16bf Af = frag_load(Arow, k0, hiSel);
#pragma unroll
    for (int t = 0; t < 4; ++t) {
      v16bf Bf = frag_load(Btb + (size_t)(n0 + t * 16 + l15) * Kp, k0, hiSel);
      acc[t] = wmma_bf16(Af, Bf, acc[t]);
    }
  }
  size_t obase = (size_t)b * M * HW;
#pragma unroll
  for (int t = 0; t < 4; ++t) {
    int n = n0 + t * 16 + l15;
#pragma unroll
    for (int r = 0; r < 8; ++r) {
      int m = blockIdx.y * 64 + mw * 16 + hiSel + r;
      if (m < M) {
        float v = acc[t][r];
        if (res) v += res[obase + (size_t)m * HW + n];
        if (OUTF32) outF[obase + (size_t)m * HW + n] = v;
        else        outB[obase + (size_t)m * HW + n] = (bf16)v;
      }
    }
  }
}

// ---------------------------------------------------------------------------
// Host-side orchestration
// ---------------------------------------------------------------------------
static inline size_t al256(size_t x) { return (x + 255) & ~(size_t)255; }

extern "C" void kernel_launch(void* const* d_in, const int* in_sizes, int n_in,
                              void* d_out, int out_size, void* d_ws, size_t ws_size,
                              hipStream_t stream) {
  (void)in_sizes; (void)n_in; (void)out_size; (void)ws_size;
  const float* in1   = (const float*)d_in[0];
  const float* in2   = (const float*)d_in[1];
  const float* lnkw  = (const float*)d_in[2];
  const float* lnkb  = (const float*)d_in[3];
  const float* lnvw  = (const float*)d_in[4];
  const float* lnvb  = (const float*)d_in[5];
  const float* lnqw  = (const float*)d_in[6];
  const float* lnqb  = (const float*)d_in[7];
  const float* wq    = (const float*)d_in[8];
  const float* wk    = (const float*)d_in[9];
  const float* wv    = (const float*)d_in[10];
  const float* wpo   = (const float*)d_in[11];
  const float* temp  = (const float*)d_in[12];
  const float* a1    = (const float*)d_in[13];
  const float* a2    = (const float*)d_in[14];
  const float* a3    = (const float*)d_in[15];
  const float* a4    = (const float*)d_in[16];
  const float* ln2w  = (const float*)d_in[17];
  const float* ln2b  = (const float*)d_in[18];
  const float* win   = (const float*)d_in[19];
  const float* wdw3  = (const float*)d_in[20];
  const float* wdw5  = (const float*)d_in[21];
  const float* wdw31 = (const float*)d_in[22];
  const float* wdw51 = (const float*)d_in[23];
  const float* wout  = (const float*)d_in[24];

  const size_t F32_C  = (size_t)NB * C192 * HW * 4;       // 25,165,824
  const size_t BF_C   = F32_C / 2;
  const size_t BF_BIG = (size_t)NB * BIG * HW * 2;        // 66,846,720

  char* ws = (char*)d_ws;
  size_t off = 0;
  auto alloc = [&](size_t bytes) { char* p = ws + off; off = al256(off + bytes); return p; };

  float* kfea   = (float*)alloc(F32_C);
  float* vfea   = (float*)alloc(F32_C);
  float* qfea   = (float*)alloc(F32_C);
  float* qconv  = (float*)alloc(F32_C);
  float* kconv  = (float*)alloc(F32_C);
  bf16*  vt     = (bf16*) alloc((size_t)NB * HW * 256 * 2);
  bf16*  qbf    = (bf16*) alloc(BF_C);
  bf16*  kbf    = (bf16*) alloc(BF_C);
  float* ssqp_q = (float*)alloc((size_t)384 * 64 * 4);
  float* ssqp_k = (float*)alloc((size_t)384 * 64 * 4);
  float* inv_q  = (float*)alloc(384 * 4);
  float* inv_k  = (float*)alloc(384 * 4);
  float* attnP  = (float*)alloc((size_t)32 * 8 * 2304 * 4);
  float* attnB  = (float*)alloc((size_t)8 * 2304 * 4);
  bf16*  Pbf    = (bf16*) alloc((size_t)8 * 48 * 64 * 2);
  bf16*  pot    = (bf16*) alloc((size_t)NB * HW * C192 * 2);
  float* xbuf   = (float*)alloc(F32_C);
  bf16*  yt     = (bf16*) alloc((size_t)NB * HW * C192 * 2);
  bf16*  tbuf   = (bf16*) alloc(BF_BIG);
  bf16*  x3     = (bf16*) alloc(BF_BIG);
  bf16*  x5     = (bf16*) alloc(BF_BIG);
  bf16*  ffnt   = (bf16*) alloc((size_t)NB * HW * BIGP * 2);
  bf16*  wpo_bf = (bf16*) alloc((size_t)192 * 192 * 2);
  bf16*  win_bf = (bf16*) alloc((size_t)1024 * 192 * 2);
  bf16*  wout_bf= (bf16*) alloc((size_t)192 * 1024 * 2);

  // ---- LayerNorms ----
  ln_dual_kernel<<<128, 256, 0, stream>>>(in1, lnkw, lnkb, lnvw, lnvb, kfea, vfea);
  ln_dual_kernel<<<128, 256, 0, stream>>>(in2, lnqw, lnqb, lnqw, lnqb, qfea, nullptr);

  // ---- depthwise q/k/v (+ l2-norm partials, v transposed bf16) ----
  dim3 gconv(64, C192, NB);
  dw3_f32_ssq<<<gconv, 256, 0, stream>>>(qfea, wq, qconv, ssqp_q);
  dw3_f32_ssq<<<gconv, 256, 0, stream>>>(kfea, wk, kconv, ssqp_k);
  dw3_vt<<<gconv, 256, 0, stream>>>(vfea, wv, vt);
  zero_vt_pad<<<(NB * HW * 64 + 255) / 256, 256, 0, stream>>>(vt);

  // ---- l2 normalization -> bf16 q,k ----
  reduce_inv<<<2, 256, 0, stream>>>(ssqp_q, inv_q, 384);
  reduce_inv<<<2, 256, 0, stream>>>(ssqp_k, inv_k, 384);
  int totC = NB * C192 * HW;
  scale_bf16_kernel<<<(totC + 255) / 256, 256, 0, stream>>>(qconv, inv_q, qbf, totC);
  scale_bf16_kernel<<<(totC + 255) / 256, 256, 0, stream>>>(kconv, inv_k, kbf, totC);

  // ---- attn logits (WMMA, split-K) + combined top-k softmax ----
  attn_gemm<<<dim3(8, 32), 32, 0, stream>>>(qbf, kbf, attnP);
  reduce_attn<<<(8 * 2304 + 255) / 256, 256, 0, stream>>>(attnP, attnB);
  combine_topk<<<8, 64, 0, stream>>>(attnB, temp, a1, a2, a3, a4, Pbf);

  // ---- P @ v (WMMA) -> transposed bf16 ----
  pv_gemm<<<dim3(HW / 64, 8), 128, 0, stream>>>(Pbf, vt, pot);

  // ---- weight packing ----
  pack_w_kernel<<<(192 * 192 + 255) / 256, 256, 0, stream>>>(wpo, wpo_bf, 192, 192, 192, 192);
  pack_w_kernel<<<(1024 * 192 + 255) / 256, 256, 0, stream>>>(win, win_bf, BIG, 192, 1024, 192);
  pack_w_kernel<<<(192 * 1024 + 255) / 256, 256, 0, stream>>>(wout, wout_bf, 192, BIG, 192, 1024);

  // ---- w_po projection + residual -> x ----
  gemm_tn<1><<<dim3(HW / 128, 3, NB), 256, 0, stream>>>(wpo_bf, pot, 192, 192, in2, xbuf, nullptr);

  // ---- FFN ----
  ln_t_kernel<<<128, 256, 0, stream>>>(xbuf, ln2w, ln2b, yt);
  gemm_tn<0><<<dim3(HW / 128, 16, NB), 256, 0, stream>>>(win_bf, yt, BIG, 192, nullptr, nullptr, tbuf);

  dim3 gbig(64, BIG, NB);
  dw_relu_bf16<3><<<gbig, 256, 0, stream>>>(tbuf, wdw3, x3);
  dw_relu_bf16<5><<<gbig, 256, 0, stream>>>(tbuf, wdw5, x5);

  dim3 ggrp(64, HALF_BIG, NB);
  grp2_bf16<3><<<ggrp, 256, 0, stream>>>(x3, x5, wdw31, ffnt, 0);
  grp2_bf16<5><<<ggrp, 256, 0, stream>>>(x3, x5, wdw51, ffnt, 1);
  zero_ffnt_pad<<<(NB * HW * 4 + 255) / 256, 256, 0, stream>>>(ffnt);

  // ---- w_out + residual x -> d_out ----
  gemm_tn<1><<<dim3(HW / 128, 3, NB), 256, 0, stream>>>(wout_bf, ffnt, 192, 1024, xbuf,
                                                        (float*)d_out, nullptr);
}
